// IFHMM_27805618274978
// MI455X (gfx1250) — compile-verified, matrix operand order checked by reading, not run on hardware
//
#include <hip/hip_runtime.h>
#include <math.h>

typedef __attribute__((ext_vector_type(16))) _Float16 v16h;
typedef __attribute__((ext_vector_type(8)))  float    v8f;

#define C_DIM 64
#define S_DIM 256
#define N_DIM 2000
#define NPAD  2048
#define T_DIM 500

// d_out layout (flat float32, reference return order)
#define OBS0_OFF 0                                  // log_obs    [T][N]
#define OBS1_OFF (T_DIM*N_DIM)                      // log_obs_   [T][C][N]
#define HID_OFF  (T_DIM*N_DIM + T_DIM*C_DIM*N_DIM)  // log_hidden [T][C][S]

// ---------------- preprocessing ----------------

// row log_softmax, row length 256, one block per row
__global__ void k_row_lsm256(const float* __restrict__ in, float* __restrict__ out) {
  __shared__ float red[256];
  int r = blockIdx.x, t = threadIdx.x;
  float x = in[r * 256 + t];
  red[t] = x; __syncthreads();
  for (int s = 128; s > 0; s >>= 1) { if (t < s) red[t] = fmaxf(red[t], red[t + s]); __syncthreads(); }
  float m = red[0]; __syncthreads();
  red[t] = __expf(x - m); __syncthreads();
  for (int s = 128; s > 0; s >>= 1) { if (t < s) red[t] += red[t + s]; __syncthreads(); }
  out[r * 256 + t] = x - (m + __logf(red[0]));
}

// chain weights: row log_softmax over C=64
__global__ void k_chain(const float* __restrict__ in, float* __restrict__ lcw) {
  __shared__ float red[64];
  int r = blockIdx.x, t = threadIdx.x;
  float x = in[r * C_DIM + t];
  red[t] = x; __syncthreads();
  for (int s = 32; s > 0; s >>= 1) { if (t < s) red[t] = fmaxf(red[t], red[t + s]); __syncthreads(); }
  float m = red[0]; __syncthreads();
  red[t] = __expf(x - m); __syncthreads();
  for (int s = 32; s > 0; s >>= 1) { if (t < s) red[t] += red[t + s]; __syncthreads(); }
  lcw[r * C_DIM + t] = x - (m + __logf(red[0]));
}

// emission rows: log_softmax over N, + log(mask), then renormalize (reference's double normalization)
__global__ void k_emis_rows(const float* __restrict__ in, const float* __restrict__ mask,
                            float* __restrict__ logE) {
  __shared__ float red[256];
  int r = blockIdx.x, t = threadIdx.x;
  float loc[8];
  float m = -INFINITY;
  for (int i = 0; i < 8; ++i) {
    int n = t + i * 256;
    loc[i] = (n < N_DIM) ? in[r * N_DIM + n] : -INFINITY;
    m = fmaxf(m, loc[i]);
  }
  red[t] = m; __syncthreads();
  for (int s = 128; s > 0; s >>= 1) { if (t < s) red[t] = fmaxf(red[t], red[t + s]); __syncthreads(); }
  m = red[0]; __syncthreads();
  float sum = 0.f;
  for (int i = 0; i < 8; ++i) { if (t + i * 256 < N_DIM) sum += __expf(loc[i] - m); }
  red[t] = sum; __syncthreads();
  for (int s = 128; s > 0; s >>= 1) { if (t < s) red[t] += red[t + s]; __syncthreads(); }
  float l1 = m + __logf(red[0]); __syncthreads();
  // y = x - l1 + log(mask); second logsumexp normalization
  float m2 = -INFINITY;
  for (int i = 0; i < 8; ++i) {
    int n = t + i * 256;
    if (n < N_DIM) { loc[i] = loc[i] - l1 + __logf(mask[r * N_DIM + n]); m2 = fmaxf(m2, loc[i]); }
  }
  red[t] = m2; __syncthreads();
  for (int s = 128; s > 0; s >>= 1) { if (t < s) red[t] = fmaxf(red[t], red[t + s]); __syncthreads(); }
  m2 = red[0]; __syncthreads();
  float s2 = 0.f;
  for (int i = 0; i < 8; ++i) { if (t + i * 256 < N_DIM) s2 += __expf(loc[i] - m2); }
  red[t] = s2; __syncthreads();
  for (int s = 128; s > 0; s >>= 1) { if (t < s) red[t] += red[t + s]; __syncthreads(); }
  float l2 = m2 + __logf(red[0]);
  for (int i = 0; i < 8; ++i) {
    int n = t + i * 256;
    if (n < N_DIM) logE[r * N_DIM + n] = loc[i] - l2;
  }
}

// column max over S + transposed f16 exp matrix, padded to NPAD columns
__global__ void k_emis_cols(const float* __restrict__ logE, float* __restrict__ bmaxE,
                            _Float16* __restrict__ expET) {
  int n = blockIdx.x * 128 + threadIdx.x;
  if (n < N_DIM) {
    float m = -INFINITY;
    for (int s = 0; s < S_DIM; ++s) m = fmaxf(m, logE[s * N_DIM + n]);
    bmaxE[n] = m;
    for (int s = 0; s < S_DIM; ++s) expET[(size_t)n * S_DIM + s] = (_Float16)__expf(logE[s * N_DIM + n] - m);
  } else {
    bmaxE[n] = 0.f;
    for (int s = 0; s < S_DIM; ++s) expET[(size_t)n * S_DIM + s] = (_Float16)0.f;
  }
}

__global__ void k_trans_cols(const float* __restrict__ logTm, float* __restrict__ bmaxT,
                             _Float16* __restrict__ expTmT) {
  int j = blockIdx.x * 128 + threadIdx.x;
  float m = -INFINITY;
  for (int i = 0; i < S_DIM; ++i) m = fmaxf(m, logTm[i * S_DIM + j]);
  bmaxT[j] = m;
  for (int i = 0; i < S_DIM; ++i) expTmT[j * S_DIM + i] = (_Float16)__expf(logTm[i * S_DIM + j] - m);
}

// ---------------- sequential recurrence (one persistent workgroup) ----------------
// h_{t} = log(exp(h_{t-1}-amax) @ exp(logTm-bmaxT)) + amax + bmaxT, written to log_hidden[t]
__global__ void __launch_bounds__(512) k_recur(float* __restrict__ out,
                                               const float* __restrict__ bmaxT,
                                               const _Float16* __restrict__ expTmT) {
  extern __shared__ char smem[];
  float*    hcur = (float*)smem;                               // 64*256 f32 = 64KB
  _Float16* expA = (_Float16*)(smem + C_DIM * S_DIM * 4);      // 64*256 f16 = 32KB
  float*    amax = (float*)(smem + C_DIM * S_DIM * 6);         // 64
  float*    part = amax + 64;                                  // 64*8

  int tid = threadIdx.x;
  int lane = tid & 31, wv = tid >> 5;     // wave32: 16 waves
  int lm = lane & 15, lg = lane >> 4;

  float* hid = out + HID_OFF;
  for (int i = tid; i < C_DIM * S_DIM; i += 512) hcur[i] = hid[i];  // h0 = log_state_init
  __syncthreads();

  for (int t = 1; t < T_DIM; ++t) {
    { // row max, 8 threads per row
      int r = tid >> 3, p = tid & 7;
      float m = -INFINITY;
      for (int s = p * 32; s < p * 32 + 32; ++s) m = fmaxf(m, hcur[r * S_DIM + s]);
      part[r * 8 + p] = m;
    }
    __syncthreads();
    if (tid < 64) {
      float m = part[tid * 8];
      for (int p = 1; p < 8; ++p) m = fmaxf(m, part[tid * 8 + p]);
      amax[tid] = m;
    }
    __syncthreads();
    for (int i = tid; i < C_DIM * S_DIM; i += 512)
      expA[i] = (_Float16)__expf(hcur[i] - amax[i >> 8]);
    __syncthreads();

    float* hout = hid + (size_t)t * C_DIM * S_DIM;
    // output 64x256 = 4x16 tiles of 16x16; each wave does 4 tiles, K loop = 8 WMMAs
    for (int tt = wv; tt < 64; tt += 16) {
      int m0 = (tt >> 4) * 16, n0 = (tt & 15) * 16;
      v8f acc = {};
      for (int kk = 0; kk < 8; ++kk) {
        int kb = kk * 32;
        v16h a, b;
#pragma unroll
        for (int e = 0; e < 16; ++e) {
          int v = e >> 1;
          int k = kb + ((v >> 2) << 4) + (lg << 3) + ((v & 3) << 1) + (e & 1);
          a[e] = expA[(m0 + lm) * S_DIM + k];
          b[e] = expTmT[(n0 + lm) * S_DIM + k];
        }
        acc = __builtin_amdgcn_wmma_f32_16x16x32_f16(false, a, false, b, (short)0, acc, false, false);
      }
#pragma unroll
      for (int i = 0; i < 8; ++i) {
        int m = m0 + i + (lg << 3);
        int n = n0 + lm;
        float v = __logf(acc[i]) + amax[m] + bmaxT[n];
        hout[m * S_DIM + n] = v;
        hcur[m * S_DIM + n] = v;
      }
    }
    __syncthreads();
  }
}

// ---------------- emission: log_obs_ and log_obs ----------------
__global__ void __launch_bounds__(256) k_emis(float* __restrict__ out,
                                              const float* __restrict__ lcw,
                                              const float* __restrict__ bmaxE,
                                              const _Float16* __restrict__ expET) {
  extern __shared__ char smem[];
  _Float16* expA = (_Float16*)smem;                 // 32768 B
  float*    red  = (float*)(smem + 32768);          // 64*128 f32 = 32768 B
  float*    amax = (float*)(smem + 65536);          // 64
  float*    part = amax + 64;                       // 64*4
  float*    slcw = part + 256;                      // 64

  int t = blockIdx.y;
  int n0base = blockIdx.x * 128;
  int tid = threadIdx.x, lane = tid & 31, wv = tid >> 5;  // 8 waves
  int lm = lane & 15, lg = lane >> 4;

  const float* hrow = out + HID_OFF + (size_t)t * C_DIM * S_DIM;
  { // row max, 4 threads per row
    int r = tid >> 2, p = tid & 3;
    float m = -INFINITY;
    for (int s = p * 64; s < p * 64 + 64; ++s) m = fmaxf(m, hrow[r * S_DIM + s]);
    part[r * 4 + p] = m;
  }
  if (tid < 64) slcw[tid] = lcw[t * C_DIM + tid];
  __syncthreads();
  if (tid < 64)
    amax[tid] = fmaxf(fmaxf(part[tid * 4], part[tid * 4 + 1]),
                      fmaxf(part[tid * 4 + 2], part[tid * 4 + 3]));
  __syncthreads();
  for (int i = tid; i < C_DIM * S_DIM; i += 256)
    expA[i] = (_Float16)__expf(hrow[i] - amax[i >> 8]);
  __syncthreads();

  // 64x128 output: 4x8 tiles; each wave does 4 tiles
  for (int tt = wv; tt < 32; tt += 8) {
    int m0 = (tt >> 3) * 16, nloc0 = (tt & 7) * 16;
    int n0 = n0base + nloc0;
    v8f acc = {};
    for (int kk = 0; kk < 8; ++kk) {
      int kb = kk * 32;
      v16h a, b;
#pragma unroll
      for (int e = 0; e < 16; ++e) {
        int v = e >> 1;
        int k = kb + ((v >> 2) << 4) + (lg << 3) + ((v & 3) << 1) + (e & 1);
        a[e] = expA[(m0 + lm) * S_DIM + k];
        b[e] = expET[(size_t)(n0 + lm) * S_DIM + k];
      }
      acc = __builtin_amdgcn_wmma_f32_16x16x32_f16(false, a, false, b, (short)0, acc, false, false);
    }
#pragma unroll
    for (int i = 0; i < 8; ++i) {
      int m = m0 + i + (lg << 3);
      int nl = nloc0 + lm;
      int n = n0base + nl;
      float v = __logf(acc[i]) + amax[m] + bmaxE[n];
      if (n < N_DIM) out[OBS1_OFF + ((size_t)t * C_DIM + m) * N_DIM + n] = v;
      red[m * 128 + nl] = v + slcw[m];   // padded cols -> -inf, masked below
    }
  }
  __syncthreads();
  if (tid < 128) {
    int n = n0base + tid;
    if (n < N_DIM) {
      float m = -INFINITY;
      for (int c = 0; c < C_DIM; ++c) m = fmaxf(m, red[c * 128 + tid]);
      float s = 0.f;
      for (int c = 0; c < C_DIM; ++c) s += __expf(red[c * 128 + tid] - m);
      out[(size_t)t * N_DIM + n] = m + __logf(s);
    }
  }
}

extern "C" void kernel_launch(void* const* d_in, const int* in_sizes, int n_in,
                              void* d_out, int out_size, void* d_ws, size_t ws_size,
                              hipStream_t stream) {
  (void)in_sizes; (void)n_in; (void)out_size; (void)ws_size;
  const float* in_si = (const float*)d_in[0];  // [C,S]
  const float* in_cw = (const float*)d_in[1];  // [T,C]
  const float* in_em = (const float*)d_in[2];  // [S,N]
  const float* in_tm = (const float*)d_in[3];  // [S,S]
  const float* in_mk = (const float*)d_in[4];  // [S,N]
  float* out = (float*)d_out;

  // workspace layout (floats then halves), ~3.5 MB
  float* wsf   = (float*)d_ws;
  float* lcw   = wsf;                   // 32000
  float* logE  = wsf + 32000;           // 512000
  float* bmaxE = wsf + 544000;          // 2048
  float* logTm = wsf + 546048;          // 65536
  float* bmaxT = wsf + 611584;          // 256
  _Float16* expET  = (_Float16*)(wsf + 611840);  // NPAD*256 halves
  _Float16* expTmT = expET + (size_t)NPAD * S_DIM;

  // preprocessing
  k_row_lsm256<<<C_DIM, 256, 0, stream>>>(in_si, out + HID_OFF);  // log_hidden[0]
  k_row_lsm256<<<S_DIM, 256, 0, stream>>>(in_tm, logTm);
  k_chain<<<T_DIM, C_DIM, 0, stream>>>(in_cw, lcw);
  k_emis_rows<<<S_DIM, 256, 0, stream>>>(in_em, in_mk, logE);
  k_emis_cols<<<NPAD / 128, 128, 0, stream>>>(logE, bmaxE, expET);
  k_trans_cols<<<S_DIM / 128, 128, 0, stream>>>(logTm, bmaxT, expTmT);

  // sequential scan: one persistent WG, LDS-resident state
  size_t smem_recur = (size_t)C_DIM * S_DIM * 6 + (64 + 512) * 4;  // 100608 B
  k_recur<<<1, 512, smem_recur, stream>>>(out, bmaxT, expTmT);

  // big emission matmul + logsumexp over C
  size_t smem_emis = 65536 + (64 + 256 + 64) * 4;  // 67072 B
  k_emis<<<dim3(NPAD / 128, T_DIM), 256, smem_emis, stream>>>(out, lcw, bmaxE, expET);
}
